// VoteFusionModuleDirectAssign_26688926777517
// MI455X (gfx1250) — matrix operand order repaired and checked by reading, not compiled, not run on hardware
//
#include <hip/hip_runtime.h>
#include <math.h>

typedef __attribute__((ext_vector_type(16))) _Float16 v16h;
typedef __attribute__((ext_vector_type(8)))  _Float16 v8h;
typedef __attribute__((ext_vector_type(8)))  float    v8f;

#define NP 16384
#define NR 8192
#define R2 0.005625f  /* 0.075^2 */

// ---------------------------------------------------------------------------
// fp32 -> fp16 conversion (weights)
// ---------------------------------------------------------------------------
__global__ __launch_bounds__(256) void k_cvt_f16(const float* __restrict__ s,
                                                 _Float16* __restrict__ d, int n) {
  int i = blockIdx.x * blockDim.x + threadIdx.x;
  int stride = gridDim.x * blockDim.x;
  for (; i < n; i += stride) d[i] = (_Float16)s[i];
}

// ---------------------------------------------------------------------------
// 3-NN + gather + max-pool + concat staging + xyz copy
// ---------------------------------------------------------------------------
__global__ __launch_bounds__(256) void k_nn_gather(
    const float* __restrict__ pcd_xyz, const float* __restrict__ rgb_xyz,
    const float* __restrict__ pcd_feat, const float* __restrict__ rgb_feat,
    _Float16* __restrict__ xcat,    // [NP][384]
    _Float16* __restrict__ x2buf,   // [NP][256] -> write ch 128..255
    _Float16* __restrict__ fullbuf, // [NP][160] -> write ch 0..31
    float* __restrict__ out_xyz)    // [NP][3]
{
  __shared__ float sx[256], sy[256], sz[256];
  int p = blockIdx.x * 256 + threadIdx.x;
  float px = pcd_xyz[p * 3 + 0], py = pcd_xyz[p * 3 + 1], pz = pcd_xyz[p * 3 + 2];
  float b0 = 1e30f, b1 = 1e30f, b2 = 1e30f;
  int i0 = 0, i1 = 0, i2 = 0;
  for (int t0 = 0; t0 < NR; t0 += 256) {
    int j = t0 + threadIdx.x;
    sx[threadIdx.x] = rgb_xyz[j * 3 + 0];
    sy[threadIdx.x] = rgb_xyz[j * 3 + 1];
    sz[threadIdx.x] = rgb_xyz[j * 3 + 2];
    __syncthreads();
    for (int jj = 0; jj < 256; ++jj) {
      float dx = px - sx[jj], dy = py - sy[jj], dz = pz - sz[jj];
      float d = dx * dx + dy * dy + dz * dz;
      int idx = t0 + jj;
      if (d < b2) {
        if (d < b1) {
          if (d < b0) { b2 = b1; i2 = i1; b1 = b0; i1 = i0; b0 = d; i0 = idx; }
          else        { b2 = b1; i2 = i1; b1 = d; i1 = idx; }
        } else        { b2 = d; i2 = idx; }
      }
    }
    __syncthreads();
  }
  bool v0 = (b0 <= R2), v1 = (b1 <= R2), v2 = (b2 <= R2);
  _Float16* xc = xcat + (size_t)p * 384;
  _Float16* mx = x2buf + (size_t)p * 256 + 128;
  for (int c = 0; c < 128; ++c) {
    float f0 = v0 ? rgb_feat[c * NR + i0] : 0.f;
    float f1 = v1 ? rgb_feat[c * NR + i1] : 0.f;
    float f2 = v2 ? rgb_feat[c * NR + i2] : 0.f;
    xc[c]       = (_Float16)f0;
    xc[128 + c] = (_Float16)f1;
    xc[256 + c] = (_Float16)f2;
    mx[c] = (_Float16)fmaxf(f0, fmaxf(f1, f2));
  }
  _Float16* fb = fullbuf + (size_t)p * 160;
  for (int c = 0; c < 32; ++c) fb[c] = (_Float16)pcd_feat[c * NP + p];
  out_xyz[p * 3 + 0] = px; out_xyz[p * 3 + 1] = py; out_xyz[p * 3 + 2] = pz;
}

// ---------------------------------------------------------------------------
// WMMA GEMM with async-to-LDS staged B tile.
//   Block = 8 waves: covers up to 128 output channels x 64 points.
//   B tile (64 points x K ch, f16) staged ONCE per block into LDS via
//   GLOBAL_LOAD_ASYNC_TO_LDS_B128 (ASYNCcnt), shared by all 8 waves.
//   LDS row stride padded to K+8 halves -> 4-dword bank advance per row.
//   W: f16 [M][K] row-major; X: f16 [N][ldx] @xoff; Y: f16 [N][ldy] @yoff.
// ---------------------------------------------------------------------------
__global__ __launch_bounds__(256) void k_gemm_wmma(
    const _Float16* __restrict__ W, const _Float16* __restrict__ X,
    _Float16* __restrict__ Y,
    const float* __restrict__ bias, const float* __restrict__ bn,
    int M, int K, int ldx, int xoff, int ldy, int yoff, int relu)
{
  extern __shared__ _Float16 smem[];   // [64][K+8]
  const int lane = threadIdx.x & 31;
  const int wave = threadIdx.x >> 5;
  const int row  = lane & 15;
  const int hi   = lane >> 4;

  const int n0     = blockIdx.x << 6;          // 64-point tile
  const int tilesM = M >> 4;
  const int mt     = blockIdx.y * 8 + wave;    // this wave's 16-channel tile
  const bool active = (mt < tilesM);
  const int m0     = mt << 4;
  const int ldk    = K + 8;

  // ---- stage B tile into LDS (all threads participate) ----
  {
    const int cols8 = K >> 3;                  // 8 halves (16B) per transfer
    const int total = cols8 << 6;              // 64 rows
    const uint32_t sbase = (uint32_t)(uintptr_t)smem;
    for (int t = threadIdx.x; t < total; t += 256) {
      int r  = t / cols8;
      int c8 = t - r * cols8;
      uint64_t g = (uint64_t)(uintptr_t)(X + (size_t)xoff +
                                         (size_t)(n0 + r) * ldx + c8 * 8);
      uint32_t l = sbase + (uint32_t)((r * ldk + c8 * 8) * 2);
      asm volatile("global_load_async_to_lds_b128 %0, %1, off"
                   :: "v"(l), "v"(g) : "memory");
    }
    asm volatile("s_wait_asynccnt 0x0" ::: "memory");
    __syncthreads();
  }

  if (active) {
    v8f acc[4];
    #pragma unroll
    for (int t = 0; t < 4; ++t)
      acc[t] = (v8f){0.f, 0.f, 0.f, 0.f, 0.f, 0.f, 0.f, 0.f};

    // A frag: lanes 0-15 -> K {0..7,16..23}; lanes 16-31 -> K {8..15,24..31}
    const _Float16* arow = W + (size_t)(m0 + row) * K + hi * 8;
    // B frag: lane holds LDS row (t*16+row); lanes 0-15 -> K {0..7,16..23}; lanes 16-31 -> K {8..15,24..31}
    const _Float16* brow = smem + row * ldk + hi * 8;

    for (int k0 = 0; k0 < K; k0 += 32) {
      v8h alo = *(const v8h*)(arow + k0);
      v8h ahi = *(const v8h*)(arow + k0 + 16);
      v16h a;
      #pragma unroll
      for (int i = 0; i < 8; ++i) { a[i] = alo[i]; a[8 + i] = ahi[i]; }
      #pragma unroll
      for (int t = 0; t < 4; ++t) {
        const _Float16* bp = brow + (t * 16) * ldk + k0;
        v8h blo = *(const v8h*)bp;
        v8h bhi = *(const v8h*)(bp + 16);
        v16h b;
        #pragma unroll
        for (int i = 0; i < 8; ++i) { b[i] = blo[i]; b[8 + i] = bhi[i]; }
        acc[t] = __builtin_amdgcn_wmma_f32_16x16x32_f16(
            false, a, false, b, (short)0, acc[t], false, false);
      }
    }

    // Epilogue: bias (+BN) (+ReLU); 8 contiguous output channels per lane.
    #pragma unroll
    for (int t = 0; t < 4; ++t) {
      const int n = n0 + t * 16 + row;
      _Float16* yp = Y + (size_t)yoff + (size_t)n * ldy + m0 + hi * 8;
      v8h o;
      #pragma unroll
      for (int r = 0; r < 8; ++r) {
        const int m = m0 + hi * 8 + r;
        float v = acc[t][r] + bias[m];
        if (bn) {
          float g = bn[m], be = bn[M + m], mu = bn[2 * M + m], va = bn[3 * M + m];
          v = (v - mu) * g * rsqrtf(va + 1e-5f) + be;
        }
        if (relu) v = v > 0.f ? v : 0.f;
        o[r] = (_Float16)v;
      }
      *(v8h*)yp = o;
    }
  }
}

// ---------------------------------------------------------------------------
// Score head tail: sigmoid(W2 . s + b), wave-per-point
// ---------------------------------------------------------------------------
__global__ __launch_bounds__(256) void k_score(const _Float16* __restrict__ act,
                                               const float* __restrict__ w,
                                               const float* __restrict__ b,
                                               float* __restrict__ out) {
  int wave = threadIdx.x >> 5, lane = threadIdx.x & 31;
  int p = blockIdx.x * 8 + wave;
  const _Float16* ap = act + (size_t)p * 128;
  float s = 0.f;
  #pragma unroll
  for (int i = 0; i < 4; ++i) { int c = lane + 32 * i; s += (float)ap[c] * w[c]; }
  #pragma unroll
  for (int off = 16; off; off >>= 1) s += __shfl_xor(s, off, 32);
  if (lane == 0) out[p] = 1.f / (1.f + expf(-(s + b[0])));
}

// ---------------------------------------------------------------------------
// L2-normalized vote features, wave-per-point
// ---------------------------------------------------------------------------
__global__ __launch_bounds__(256) void k_vf(const _Float16* __restrict__ fp,
                                            float* __restrict__ out) {
  int wave = threadIdx.x >> 5, lane = threadIdx.x & 31;
  int p = blockIdx.x * 8 + wave;
  const _Float16* ap = fp + (size_t)p * 128;
  float vals[4], ss = 0.f;
  #pragma unroll
  for (int i = 0; i < 4; ++i) { vals[i] = (float)ap[lane + 32 * i]; ss += vals[i] * vals[i]; }
  #pragma unroll
  for (int off = 16; off; off >>= 1) ss += __shfl_xor(ss, off, 32);
  float inv = 1.f / fmaxf(sqrtf(ss), 1e-12f);
  #pragma unroll
  for (int i = 0; i < 4; ++i) out[(size_t)p * 128 + lane + 32 * i] = vals[i] * inv;
}

// ---------------------------------------------------------------------------
extern "C" void kernel_launch(void* const* d_in, const int* in_sizes, int n_in,
                              void* d_out, int out_size, void* d_ws, size_t ws_size,
                              hipStream_t stream) {
  const float* pcd_xyz  = (const float*)d_in[0];
  const float* rgb_xyz  = (const float*)d_in[1];
  const float* pcd_feat = (const float*)d_in[2];
  const float* rgb_feat = (const float*)d_in[3];
  const float* cc1_w = (const float*)d_in[4];  const float* cc1_b = (const float*)d_in[5];
  const float* cc_bn = (const float*)d_in[6];
  const float* cc2_w = (const float*)d_in[7];  const float* cc2_b = (const float*)d_in[8];
  const float* co1_w = (const float*)d_in[9];  const float* co1_b = (const float*)d_in[10];
  const float* co_bn = (const float*)d_in[11];
  const float* co2_w = (const float*)d_in[12]; const float* co2_b = (const float*)d_in[13];
  const float* dh1_w = (const float*)d_in[14]; const float* dh1_b = (const float*)d_in[15];
  const float* dh1_bn = (const float*)d_in[16];
  const float* dh2_w = (const float*)d_in[17]; const float* dh2_b = (const float*)d_in[18];
  const float* dh2_bn = (const float*)d_in[19];
  const float* dh3_w = (const float*)d_in[20]; const float* dh3_b = (const float*)d_in[21];
  const float* sh1_w = (const float*)d_in[22]; const float* sh1_b = (const float*)d_in[23];
  const float* sh_bn = (const float*)d_in[24];
  const float* sh2_w = (const float*)d_in[25]; const float* sh2_b = (const float*)d_in[26];

  // workspace carve-up
  char* base = (char*)d_ws;
  size_t off = 0;
  auto carve = [&](size_t bytes) -> char* {
    char* p = base + off;
    off = (off + bytes + 255) & ~(size_t)255;
    return p;
  };
  _Float16* bufA = (_Float16*)carve((size_t)NP * 384 * 2);
  _Float16* bufB = (_Float16*)carve((size_t)NP * 384 * 2);
  _Float16* bufC = (_Float16*)carve((size_t)NP * 256 * 2);
  _Float16* bufD = (_Float16*)carve((size_t)NP * 160 * 2);
  _Float16* hcc1 = (_Float16*)carve(384 * 384 * 2);
  _Float16* hcc2 = (_Float16*)carve(128 * 384 * 2);
  _Float16* hco1 = (_Float16*)carve(256 * 256 * 2);
  _Float16* hco2 = (_Float16*)carve(128 * 256 * 2);
  _Float16* hdh1 = (_Float16*)carve(160 * 160 * 2);
  _Float16* hdh2 = (_Float16*)carve(160 * 160 * 2);
  _Float16* hdh3 = (_Float16*)carve(128 * 160 * 2);
  _Float16* hsh1 = (_Float16*)carve(128 * 128 * 2);

  float* out_xyz    = (float*)d_out;               // 16384*3
  float* out_scores = (float*)d_out + NP * 3;      // 16384
  float* out_vf     = (float*)d_out + NP * 4;      // 16384*128

  // 1) weight conversion fp32 -> fp16
  auto cvt = [&](const float* s, _Float16* d, int n) {
    k_cvt_f16<<<(n + 255) / 256, 256, 0, stream>>>(s, d, n);
  };
  cvt(cc1_w, hcc1, 384 * 384); cvt(cc2_w, hcc2, 128 * 384);
  cvt(co1_w, hco1, 256 * 256); cvt(co2_w, hco2, 128 * 256);
  cvt(dh1_w, hdh1, 160 * 160); cvt(dh2_w, hdh2, 160 * 160);
  cvt(dh3_w, hdh3, 128 * 160); cvt(sh1_w, hsh1, 128 * 128);

  // 2) 3-NN + gather + concat staging (+ xyz passthrough)
  k_nn_gather<<<NP / 256, 256, 0, stream>>>(pcd_xyz, rgb_xyz, pcd_feat, rgb_feat,
                                            bufA, bufC, bufD, out_xyz);

  // 3) GEMM chain (WMMA + async-LDS staged B)
  auto gemm = [&](const _Float16* W, const _Float16* X, _Float16* Y,
                  const float* bias, const float* bn,
                  int M, int K, int ldx, int xoff, int ldy, int yoff, int relu) {
    dim3 grid(NP / 64, ((M >> 4) + 7) / 8);
    size_t shmem = (size_t)64 * (K + 8) * 2;
    k_gemm_wmma<<<grid, 256, shmem, stream>>>(W, X, Y, bias, bn,
                                              M, K, ldx, xoff, ldy, yoff, relu);
  };
  // cc1: xcat(384, bufA) -> bufB(384), BN+ReLU
  gemm(hcc1, bufA, bufB, cc1_b, cc_bn, 384, 384, 384, 0, 384, 0, 1);
  // cc2: bufB(384) -> bufC ch 0..127 (x2 = [concat_out | maxfeat])
  gemm(hcc2, bufB, bufC, cc2_b, nullptr, 128, 384, 384, 0, 256, 0, 0);
  // co1: x2(256, bufC) -> bufB(256), BN+ReLU
  gemm(hco1, bufC, bufB, co1_b, co_bn, 256, 256, 256, 0, 256, 0, 1);
  // co2: bufB(256) -> bufD ch 32..159 (full = [pcd_feat | attach])
  gemm(hco2, bufB, bufD, co2_b, nullptr, 128, 256, 256, 0, 160, 32, 0);
  // dh1: full(160, bufD) -> bufA(160), BN+ReLU
  gemm(hdh1, bufD, bufA, dh1_b, dh1_bn, 160, 160, 160, 0, 160, 0, 1);
  // dh2: bufA(160) -> bufB(160), BN+ReLU
  gemm(hdh2, bufA, bufB, dh2_b, dh2_bn, 160, 160, 160, 0, 160, 0, 1);
  // dh3: bufB(160) -> bufC(128) = fp
  gemm(hdh3, bufB, bufC, dh3_b, nullptr, 128, 160, 160, 0, 128, 0, 0);
  // sh1: fp(128, bufC) -> bufA(128), BN+ReLU
  gemm(hsh1, bufC, bufA, sh1_b, sh_bn, 128, 128, 128, 0, 128, 0, 1);

  // 4) tails
  k_score<<<NP / 8, 256, 0, stream>>>(bufA, sh2_w, sh2_b, out_scores);
  k_vf<<<NP / 8, 256, 0, stream>>>(bufC, out_vf);
}
